// ParallelControllerMoE_23476291240207
// MI455X (gfx1250) — matrix operand order repaired and checked by reading, not compiled, not run on hardware
//
#include <hip/hip_runtime.h>

// ---------------------------------------------------------------------------
// MoE with per-expert LSTM controllers on gfx1250 (MI455X).
//   S=128, B=32, I=O=H=512, E=8, K=2.
// Strategy: router -> bf16 weight prep (transposed for WMMA B-fragments) ->
// 16 persistent workgroups (one per (k,expert) pair) running the recurrence
// with v_wmma_f32_16x16x32_bf16, h in LDS (bf16), c in VGPRs, masked output
// projection accumulated with deterministic f32 atomics.
// ---------------------------------------------------------------------------

typedef __bf16 bf16_t;
typedef __attribute__((ext_vector_type(8)))  bf16_t v8bf;
typedef __attribute__((ext_vector_type(16))) bf16_t v16bf;
typedef __attribute__((ext_vector_type(8)))  float  v8f;

#define S_LEN 128
#define BATCH 32
#define IDIM  512
#define HDIM  512
#define ODIM  512
#define NEXP  8
#define TOPK  2
#define NTOK  (S_LEN * BATCH)    // 4096
#define G4H   (4 * HDIM)         // 2048
#define KCAT  (IDIM + HDIM)      // 1024

// ---- workspace layout (bytes) ---------------------------------------------
static constexpr size_t WS_RIDX = 0;                                        // int   [NTOK][2]
static constexpr size_t WS_RWGT = WS_RIDX + (size_t)NTOK * TOPK * 4;        // float [NTOK][2]
static constexpr size_t WS_XBF  = WS_RWGT + (size_t)NTOK * TOPK * 4;        // bf16  [NTOK][I]
static constexpr size_t WS_WT   = WS_XBF  + (size_t)NTOK * IDIM * 2;        // bf16  [E][4H][KCAT]  (Wi||Wh transposed)
static constexpr size_t WS_WOT  = WS_WT   + (size_t)NEXP * G4H * KCAT * 2;  // bf16  [E][O][H]      (Wo transposed)

// ---- helpers --------------------------------------------------------------
__device__ __forceinline__ v8f wmma_bf16(v16bf a, v16bf b, v8f c) {
  // (neg_a, A, neg_b, B, c_mod, C, reuse_a, reuse_b)
  return __builtin_amdgcn_wmma_f32_16x16x32_bf16(false, a, false, b, (short)0, c,
                                                 false, false);
}

// A-matrix fragment, 16x32 bf16 (MxK), ISA 7.12.2 layout:
// lanes 0-15: M=lane, K chunks {k..k+7} and {k+16..k+23};
// lanes 16-31: M=lane-16, K chunks {k+8..k+15} and {k+24..k+31}.
__device__ __forceinline__ v16bf load_a_frag(const bf16_t* base, int ld, int mstart,
                                             int kk, int lane) {
  const int m    = lane & 15;
  const int hsel = lane >> 4;
  const bf16_t* p = base + (size_t)(mstart + m) * ld + kk + (hsel << 3);
  v8bf lo = *(const v8bf*)p;
  v8bf hi = *(const v8bf*)(p + 16);
  return __builtin_shufflevector(lo, hi, 0, 1, 2, 3, 4, 5, 6, 7,
                                 8, 9, 10, 11, 12, 13, 14, 15);
}

// B-matrix fragment, 32x16 bf16 (KxN), from N-major (transposed) weights:
// lanes 0-15: N=lane, K chunks {k..k+7} and {k+16..k+23};
// lanes 16-31: N=lane-16, K chunks {k+8..k+15} and {k+24..k+31}.
__device__ __forceinline__ v16bf load_b_frag(const bf16_t* base, int ld, int n0,
                                             int kk, int lane) {
  const int n    = lane & 15;
  const int hsel = lane >> 4;
  const bf16_t* p = base + (size_t)(n0 + n) * ld + kk + (hsel << 3);
  v8bf lo = *(const v8bf*)p;
  v8bf hi = *(const v8bf*)(p + 16);
  return __builtin_shufflevector(lo, hi, 0, 1, 2, 3, 4, 5, 6, 7,
                                 8, 9, 10, 11, 12, 13, 14, 15);
}

__device__ __forceinline__ float fsigmoid(float x) {
  return 1.0f / (1.0f + __expf(-x));
}
__device__ __forceinline__ float ftanh(float x) {
  // 1 - 2/(e^{2x}+1): saturates correctly to +-1 for large |x|
  return 1.0f - 2.0f / (__expf(2.0f * x) + 1.0f);
}

// ---- prolog kernels -------------------------------------------------------
__global__ void k_zero(float* __restrict__ out, int n) {
  int i = blockIdx.x * 256 + threadIdx.x;
  if (i < n) out[i] = 0.0f;
}

__global__ void k_router(const float* __restrict__ x, const float* __restrict__ rw,
                         const float* __restrict__ rb, int* __restrict__ ridx,
                         float* __restrict__ rwgt) {
  int tok = blockIdx.x * 64 + threadIdx.x;
  if (tok >= NTOK) return;
  float lg[NEXP];
#pragma unroll
  for (int e = 0; e < NEXP; ++e) lg[e] = rb[e];
  const float* xp = x + (size_t)tok * IDIM;
  for (int i = 0; i < IDIM; ++i) {
    float xv = xp[i];
#pragma unroll
    for (int e = 0; e < NEXP; ++e) lg[e] += xv * rw[i * NEXP + e];
  }
  // top-2 with first-index-on-tie (matches jax.lax.top_k)
  int i0 = 0; float v0 = lg[0];
#pragma unroll
  for (int e = 1; e < NEXP; ++e) if (lg[e] > v0) { v0 = lg[e]; i0 = e; }
  int i1 = (i0 == 0) ? 1 : 0; float v1 = lg[i1];
#pragma unroll
  for (int e = 0; e < NEXP; ++e)
    if (e != i0 && lg[e] > v1) { v1 = lg[e]; i1 = e; }
  float e1 = __expf(v1 - v0);        // e0 == 1
  float inv = 1.0f / (1.0f + e1);
  ridx[tok * 2 + 0] = i0; ridx[tok * 2 + 1] = i1;
  rwgt[tok * 2 + 0] = inv; rwgt[tok * 2 + 1] = e1 * inv;
}

__global__ void k_xbf(const float* __restrict__ x, bf16_t* __restrict__ xb, int n) {
  int i = blockIdx.x * 256 + threadIdx.x;
  if (i < n) xb[i] = (bf16_t)x[i];
}

// Wt[e][n][kc] = (kc<512 ? Wi[e][kc][n] : Wh[e][kc-512][n]), bf16, N-major.
__global__ void k_wt(const float* __restrict__ Wi, const float* __restrict__ Wh,
                     bf16_t* __restrict__ Wt) {
  size_t i = (size_t)blockIdx.x * 256 + threadIdx.x;   // E*G4H*KCAT = 16M
  int    kc = (int)(i & (KCAT - 1));
  size_t t  = i >> 10;
  int    n  = (int)(t & (G4H - 1));
  int    e  = (int)(t >> 11);
  float v = (kc < IDIM) ? Wi[((size_t)e * IDIM + kc) * G4H + n]
                        : Wh[((size_t)e * HDIM + (kc - IDIM)) * G4H + n];
  Wt[i] = (bf16_t)v;
}

// Wot[e][o][h] = Wo[e][h][o], bf16, O-major.
__global__ void k_wot(const float* __restrict__ Wo, bf16_t* __restrict__ Wot) {
  size_t i = (size_t)blockIdx.x * 256 + threadIdx.x;   // E*O*H = 2M
  int    h = (int)(i & (HDIM - 1));
  size_t t = i >> 9;
  int    o = (int)(t & (ODIM - 1));
  int    e = (int)(t >> 9);
  Wot[i] = (bf16_t)Wo[((size_t)e * HDIM + h) * ODIM + o];
}

// ---- main persistent kernel: one workgroup per (k, expert) ----------------
__global__ __launch_bounds__(512, 1)
void k_moe_lstm(const bf16_t* __restrict__ xb,
                const bf16_t* __restrict__ Wt,
                const bf16_t* __restrict__ Wot,
                const float*  __restrict__ gbias,   // b  [E][4H]
                const float*  __restrict__ obias,   // bo [E][O]
                const int*    __restrict__ ridx,
                const float*  __restrict__ rwgt,
                float*        __restrict__ out) {
  extern __shared__ char smem[];
  bf16_t* xs = (bf16_t*)smem;                          // [32][512] masked x_t
  bf16_t* hb = (bf16_t*)(smem + BATCH * IDIM * 2);     // [32][512] h state

  const int pair  = blockIdx.x;
  const int krank = pair >> 3;   // which top-k rank
  const int e     = pair & 7;    // which expert

  const int tid  = threadIdx.x;
  const int wave = tid >> 5;     // 16 waves
  const int lane = tid & 31;

  const bf16_t* WtE = Wt  + (size_t)e * G4H * KCAT;
  const bf16_t* WoE = Wot + (size_t)e * ODIM * HDIM;
  const float*  bE  = gbias + e * G4H;
  const float*  boE = obias + e * ODIM;

  const int jw = wave * 32;      // this wave's hidden/output column slice

  // zero h state
  {
    v8bf z = {};
    v8bf* hv = (v8bf*)hb;
#pragma unroll
    for (int i = 0; i < 4; ++i) hv[tid + i * 512] = z;
  }

  // hoist per-lane biases (constant across steps)
  float biasv[2][4], bov[2];
#pragma unroll
  for (int g = 0; g < 2; ++g) {
    int col = jw + g * 16 + (lane & 15);
    biasv[g][0] = bE[col];              // i
    biasv[g][1] = bE[HDIM + col];       // f
    biasv[g][2] = bE[2 * HDIM + col];   // g
    biasv[g][3] = bE[3 * HDIM + col];   // o
    bov[g] = boE[col];
  }

  v8f cst[2][2];                 // cell state: [group][mtile], register-resident
  {
    v8f zz = {};
#pragma unroll
    for (int g = 0; g < 2; ++g) { cst[g][0] = zz; cst[g][1] = zz; }
  }

  __syncthreads();

  for (int s = 0; s < S_LEN; ++s) {
    // ---- phase 1: masked x_t into LDS (zero-padded dispatch) ----
    {
      int brow = tid >> 4;                 // 32 rows, 16 threads/row
      int seg  = tid & 15;                 // 32 halves (64B) per thread
      int tok  = s * BATCH + brow;
      bool on  = (ridx[tok * 2 + krank] == e);
      const v8bf* src = (const v8bf*)(xb + (size_t)tok * IDIM + seg * 32);
      v8bf* dst = (v8bf*)(xs + brow * IDIM + seg * 32);
      v8bf z = {};
#pragma unroll
      for (int i = 0; i < 4; ++i) dst[i] = on ? src[i] : z;
    }
    __syncthreads();

    // ---- phase 2: gates = [x|h] @ [Wi;Wh] + b, then LSTM nonlinearity ----
    float hnew[2][2][8];
#pragma unroll
    for (int g = 0; g < 2; ++g) {
      const int jb = jw + g * 16;
      v8f acc[4][2];
      {
        v8f zz = {};
#pragma unroll
        for (int gt = 0; gt < 4; ++gt) { acc[gt][0] = zz; acc[gt][1] = zz; }
      }
      // K = 0..511 : x part
#pragma unroll 2
      for (int kk = 0; kk < IDIM; kk += 32) {
        v16bf a0 = load_a_frag(xs, IDIM, 0, kk, lane);
        v16bf a1 = load_a_frag(xs, IDIM, 16, kk, lane);
#pragma unroll
        for (int gt = 0; gt < 4; ++gt) {
          v16bf bfr = load_b_frag(WtE, KCAT, gt * HDIM + jb, kk, lane);
          acc[gt][0] = wmma_bf16(a0, bfr, acc[gt][0]);
          acc[gt][1] = wmma_bf16(a1, bfr, acc[gt][1]);
        }
      }
      // K = 512..1023 : h part
#pragma unroll 2
      for (int kk = 0; kk < HDIM; kk += 32) {
        v16bf a0 = load_a_frag(hb, HDIM, 0, kk, lane);
        v16bf a1 = load_a_frag(hb, HDIM, 16, kk, lane);
#pragma unroll
        for (int gt = 0; gt < 4; ++gt) {
          v16bf bfr = load_b_frag(WtE, KCAT, gt * HDIM + jb, IDIM + kk, lane);
          acc[gt][0] = wmma_bf16(a0, bfr, acc[gt][0]);
          acc[gt][1] = wmma_bf16(a1, bfr, acc[gt][1]);
        }
      }
      // nonlinearity + state update, all in registers
#pragma unroll
      for (int mt = 0; mt < 2; ++mt) {
#pragma unroll
        for (int r = 0; r < 8; ++r) {
          float iv = fsigmoid(acc[0][mt][r] + biasv[g][0]);
          float fv = fsigmoid(acc[1][mt][r] + biasv[g][1]);
          float gv = ftanh(acc[2][mt][r] + biasv[g][2]);
          float ov = fsigmoid(acc[3][mt][r] + biasv[g][3]);
          float cv = fv * cst[g][mt][r] + iv * gv;
          cst[g][mt][r] = cv;
          hnew[g][mt][r] = ov * ftanh(cv);
        }
      }
    }
    __syncthreads();   // all reads of h_{s-1} complete

    // ---- write h_s to LDS (D-tile layout: row = mt*16 + (lane>>4)*8 + r) ----
#pragma unroll
    for (int g = 0; g < 2; ++g) {
      int col = jw + g * 16 + (lane & 15);
#pragma unroll
      for (int mt = 0; mt < 2; ++mt) {
#pragma unroll
        for (int r = 0; r < 8; ++r) {
          int row = mt * 16 + ((lane >> 4) << 3) + r;
          hb[row * HDIM + col] = (bf16_t)hnew[g][mt][r];
        }
      }
    }
    __syncthreads();   // h_s visible to all waves

    // ---- phase 3: y += rw * (h_s @ Wo + bo) at routed rows only ----
    float wrow[2][8];
#pragma unroll
    for (int mt = 0; mt < 2; ++mt) {
#pragma unroll
      for (int r = 0; r < 8; ++r) {
        int row = mt * 16 + ((lane >> 4) << 3) + r;
        int tok = s * BATCH + row;
        wrow[mt][r] = (ridx[tok * 2 + krank] == e) ? rwgt[tok * 2 + krank] : 0.0f;
      }
    }
#pragma unroll
    for (int nt = 0; nt < 2; ++nt) {
      const int o0 = jw + nt * 16;
      v8f p0 = {}, p1 = {};
#pragma unroll 2
      for (int kk = 0; kk < HDIM; kk += 32) {
        v16bf a0  = load_a_frag(hb, HDIM, 0, kk, lane);
        v16bf a1  = load_a_frag(hb, HDIM, 16, kk, lane);
        v16bf bfr = load_b_frag(WoE, HDIM, o0, kk, lane);
        p0 = wmma_bf16(a0, bfr, p0);
        p1 = wmma_bf16(a1, bfr, p1);
      }
      const int col = o0 + (lane & 15);
      const float bv = bov[nt];
#pragma unroll
      for (int mt = 0; mt < 2; ++mt) {
#pragma unroll
        for (int r = 0; r < 8; ++r) {
          float wv = wrow[mt][r];
          if (wv != 0.0f) {
            int row = mt * 16 + ((lane >> 4) << 3) + r;
            float val = (mt == 0 ? p0[r] : p1[r]) + bv;
            // exactly 2 commutative atomic adds per element -> deterministic
            atomicAdd(&out[(size_t)(s * BATCH + row) * ODIM + col], wv * val);
          }
        }
      }
    }
  }
}

// ---------------------------------------------------------------------------
extern "C" void kernel_launch(void* const* d_in, const int* in_sizes, int n_in,
                              void* d_out, int out_size, void* d_ws, size_t ws_size,
                              hipStream_t stream) {
  const float* x  = (const float*)d_in[0];
  const float* rw = (const float*)d_in[1];
  const float* rb = (const float*)d_in[2];
  const float* Wi = (const float*)d_in[3];
  const float* Wh = (const float*)d_in[4];
  const float* gb = (const float*)d_in[5];
  const float* Wo = (const float*)d_in[6];
  const float* bo = (const float*)d_in[7];
  float* out = (float*)d_out;

  char* ws = (char*)d_ws;
  int*    ridx = (int*)(ws + WS_RIDX);
  float*  rwgt = (float*)(ws + WS_RWGT);
  bf16_t* xbf  = (bf16_t*)(ws + WS_XBF);
  bf16_t* Wt   = (bf16_t*)(ws + WS_WT);
  bf16_t* Wot  = (bf16_t*)(ws + WS_WOT);

  k_zero<<<(NTOK * ODIM + 255) / 256, 256, 0, stream>>>(out, NTOK * ODIM);
  k_router<<<NTOK / 64, 64, 0, stream>>>(x, rw, rb, ridx, rwgt);
  k_xbf<<<(NTOK * IDIM + 255) / 256, 256, 0, stream>>>(x, xbf, NTOK * IDIM);
  k_wt<<<(NEXP * G4H * KCAT) / 256, 256, 0, stream>>>(Wi, Wh, Wt);
  k_wot<<<(NEXP * ODIM * HDIM) / 256, 256, 0, stream>>>(Wo, Wot);

  // 16 persistent workgroups (one per (k, expert)); 64 KB dynamic LDS.
  k_moe_lstm<<<TOPK * NEXP, 512, 2 * BATCH * HDIM * (int)sizeof(bf16_t), stream>>>(
      xbf, Wt, Wot, gb, bo, ridx, rwgt, out);
}
